// SupplyGNN_20633022890167
// MI455X (gfx1250) — matrix operand order repaired and checked by reading, not compiled, not run on hardware
//
#include <hip/hip_runtime.h>
#include <hip/hip_bf16.h>

// ---------------- problem constants (match reference) ----------------
#define NNODES 50000
#define NEDGES 800000
#define ETOT   (NEDGES + NNODES)   // edges + self loops = 850000
#define FIN    128
#define HC     64                  // HEADS*HID
#define HIDC   32
#define OUTC   16
#define NEGS   0.2f
#define BNEPS  1e-5f

typedef _Float16 v16h __attribute__((ext_vector_type(16)));
typedef _Float16 v8h  __attribute__((ext_vector_type(8)));
typedef float    v8f  __attribute__((ext_vector_type(8)));

// order-preserving float -> uint key (for atomic segment max)
__device__ __forceinline__ unsigned fkey(float f) {
    unsigned u = __float_as_uint(f);
    return (u & 0x80000000u) ? ~u : (u | 0x80000000u);
}
__device__ __forceinline__ float funkey(unsigned k) {
    unsigned u = (k & 0x80000000u) ? (k ^ 0x80000000u) : ~k;
    return __uint_as_float(u);
}
__device__ __forceinline__ float lrelu(float m) { return (m > 0.f) ? m : NEGS * m; }

// ---------------- weight pre-pack into WMMA B-fragment order ----------------
// B frag (32x16 f16): lane l holds column n = l&15; half t holds k_local = (t&7) + 8*(l>>4) + 16*(t>>3)
// dst[((kt*8+ct)*32 + l)*16 + t] = Wcat[kt*32 + klocal][ct*16 + (l&15)]
__global__ void pack_weights(int K, const float* __restrict__ Wl,
                             const float* __restrict__ Wr, _Float16* __restrict__ dst) {
    int id = blockIdx.x * blockDim.x + threadIdx.x;
    if (id >= K * 128) return;
    int t  = id & 15;
    int l  = (id >> 4) & 31;
    int ct = (id >> 9) & 7;
    int kt = id >> 12;
    int k  = kt * 32 + ((t & 7) + 8 * (l >> 4) + 16 * (t >> 3));
    int c  = ct * 16 + (l & 15);
    float v = (c < HC) ? Wl[k * HC + c] : Wr[k * HC + (c - HC)];
    dst[id] = (_Float16)v;
}

// ---------------- f32 -> f16 activation convert ----------------
__global__ void cvt_f16(const float* __restrict__ src, _Float16* __restrict__ dst, int n) {
    int i = blockIdx.x * blockDim.x + threadIdx.x;
    if (i < n) dst[i] = (_Float16)src[i];
}

// ---------------- per-layer scratch re-init (graph-replay safe) ----------------
__global__ void init_layer(float* __restrict__ accum, unsigned* __restrict__ amax,
                           float* __restrict__ denom) {
    int i = blockIdx.x * blockDim.x + threadIdx.x;
    if (i < NNODES * HC) accum[i] = 0.f;
    if (i < NNODES * 2) { amax[i] = 0u; denom[i] = 0.f; }
}

// ---------------- WMMA GEMM:  C[N x 128] = A_f16[N x K] @ Bpk + [biasL|biasR] ----
// block = 256 thr = 8 waves; block covers 128 rows; wave w -> rows [base+16w, +16)
// Bpk staged into LDS via GLOBAL_LOAD_ASYNC_TO_LDS_B128 (ASYNCcnt), WMMA fed by ds_load.
// A frag (16x32 f16): lane row m = lane&15, half t -> k_local = t + 8*((t>>3) + (lane>>4))
__global__ void gemm_wmma(const _Float16* __restrict__ A, int K,
                          const _Float16* __restrict__ Bpk,
                          const float* __restrict__ biasL, const float* __restrict__ biasR,
                          float* __restrict__ C, int nRows) {
    __shared__ _Float16 Blds[16384];            // 32 KB max (K=128)
    const int KT = K >> 5;
    const int totalBytes = KT * 8 * 512 * 2;    // = K*256 bytes
    // async DMA: memory -> LDS, no VGPR round-trip (CDNA5 async path)
    for (int i = threadIdx.x * 16; i < totalBytes; i += 256 * 16) {
        unsigned loff = (unsigned)(size_t)((char*)Blds + i);            // LDS byte addr
        unsigned long long gaddr = (unsigned long long)((const char*)Bpk + i);
        asm volatile("global_load_async_to_lds_b128 %0, %1, off"
                     :: "v"(loff), "v"(gaddr) : "memory");
    }
    asm volatile("s_wait_asynccnt 0x0" ::: "memory");
    __syncthreads();

    const int wave = threadIdx.x >> 5;
    const int lane = threadIdx.x & 31;
    const int m15 = lane & 15, hi = lane >> 4;
    const long rowBase = (long)blockIdx.x * 128 + wave * 16;

    v8f acc[8];
    v8f zf = {};
#pragma unroll
    for (int t = 0; t < 8; ++t) acc[t] = zf;

    // clamp: rows >= nRows read row nRows-1 (garbage only lands in rows never stored)
    long r = rowBase + m15;
    if (r >= nRows) r = nRows - 1;
    const _Float16* arow = A + (size_t)r * K + 8 * hi;

    for (int kt = 0; kt < KT; ++kt) {
        const _Float16* ap = arow + kt * 32;
        v8h lo  = *(const v8h*)(ap);        // k_local = 8*hi + 0..7
        v8h hi8 = *(const v8h*)(ap + 16);   // k_local = 8*hi + 16..23
        v16h a = __builtin_shufflevector(lo, hi8, 0,1,2,3,4,5,6,7,8,9,10,11,12,13,14,15);
        const _Float16* bp_ = Blds + (kt * 8) * 512 + lane * 16;
#pragma unroll
        for (int ct = 0; ct < 8; ++ct) {
            v16h b = *(const v16h*)(bp_ + ct * 512);
            acc[ct] = __builtin_amdgcn_wmma_f32_16x16x32_f16(
                false, a, false, b, (short)0, acc[ct], false, false);
        }
    }
    // C frag: lane col n = lane&15; VGPR v -> row m = v + 8*(lane>>4)
#pragma unroll
    for (int ct = 0; ct < 8; ++ct) {
        int col = ct * 16 + m15;
        float bias = (col < HC) ? biasL[col] : biasR[col - HC];
#pragma unroll
        for (int v = 0; v < 8; ++v) {
            long row = rowBase + v + 8 * hi;
            if (row < nRows) C[(size_t)row * 128 + col] = acc[ct][v] + bias;
        }
    }
}

// ---------------- edge pass 1: alpha + segment max ----------------
// wave per edge, both heads; lane handles channels 2*lane, 2*lane+1
// lanes 0-15 -> head 0, lanes 16-31 -> head 1 (half-wave reductions)
__global__ void edge_alpha(const float* __restrict__ xlxr, const int* __restrict__ ei,
                           const float* __restrict__ att,
                           float* __restrict__ alpha, unsigned* __restrict__ amax) {
    int e = (blockIdx.x * blockDim.x + threadIdx.x) >> 5;
    if (e >= ETOT) return;
    int lane = threadIdx.x & 31;
    int s, d;
    if (e < NEDGES) { s = ei[e]; d = ei[NEDGES + e]; } else { s = d = e - NEDGES; }
    __builtin_prefetch(xlxr + (size_t)s * 128, 0, 3);   // global_prefetch_b8
    float2 xlv = *(const float2*)(xlxr + (size_t)s * 128 + lane * 2);
    float2 xrv = *(const float2*)(xlxr + (size_t)d * 128 + HC + lane * 2);
    float2 at  = *(const float2*)(att + lane * 2);       // att flat over 64 channels
    float p = lrelu(xlv.x + xrv.x) * at.x + lrelu(xlv.y + xrv.y) * at.y;
#pragma unroll
    for (int o = 1; o <= 8; o <<= 1) p += __shfl_xor(p, o, 32); // per-head (16-lane) sum
    if ((lane & 15) == 0) {
        int h = lane >> 4;
        alpha[(size_t)e * 2 + h] = p;
        atomicMax(&amax[d * 2 + h], fkey(p));
    }
}

// ---------------- edge pass 2: exp + weighted scatter ----------------
__global__ void edge_accum(const float* __restrict__ xlxr, const int* __restrict__ ei,
                           const float* __restrict__ alpha, const unsigned* __restrict__ amax,
                           float* __restrict__ denom, float* __restrict__ accum) {
    int e = (blockIdx.x * blockDim.x + threadIdx.x) >> 5;
    if (e >= ETOT) return;
    int lane = threadIdx.x & 31;
    int h = lane >> 4;
    int s, d;
    if (e < NEDGES) { s = ei[e]; d = ei[NEDGES + e]; } else { s = d = e - NEDGES; }
    float a  = alpha[(size_t)e * 2 + h];
    float mx = funkey(amax[d * 2 + h]);
    float ex = __expf(a - mx);                 // uniform within each 16-lane half
    float2 xlv = *(const float2*)(xlxr + (size_t)s * 128 + lane * 2);
    float* dstp = accum + (size_t)d * HC + lane * 2;
    atomicAdd(dstp + 0, ex * xlv.x);
    atomicAdd(dstp + 1, ex * xlv.y);
    if ((lane & 15) == 0) atomicAdd(&denom[d * 2 + h], ex);
}

// ---------------- node epilogue layer 1: divide + bias + relu -> f16 -----------
__global__ void epilogue1(const float* __restrict__ accum, const float* __restrict__ denom,
                          const float* __restrict__ b, _Float16* __restrict__ h16) {
    int idx = blockIdx.x * blockDim.x + threadIdx.x;
    if (idx >= NNODES * HC) return;
    int i = idx >> 6, c = idx & 63, h = c >> 5;
    float v = accum[idx] / denom[i * 2 + h] + b[c];
    v = (v > 0.f) ? v : 0.f;
    h16[idx] = (_Float16)v;
}

// ---------------- node epilogue layer 2: bias + BN + relu -> f32 ---------------
__global__ void epilogue2(const float* __restrict__ accum, const float* __restrict__ denom,
                          const float* __restrict__ b,
                          const float* __restrict__ bng, const float* __restrict__ bnb,
                          const float* __restrict__ bnm, const float* __restrict__ bnv,
                          float* __restrict__ h2) {
    int idx = blockIdx.x * blockDim.x + threadIdx.x;
    if (idx >= NNODES * HC) return;
    int i = idx >> 6, c = idx & 63, h = c >> 5;
    float v = accum[idx] / denom[i * 2 + h] + b[c];
    v = (v - bnm[c]) * rsqrtf(bnv[c] + BNEPS) * bng[c] + bnb[c];
    v = (v > 0.f) ? v : 0.f;
    h2[idx] = v;
}

// ---------------- projection + risk head: wave per node ----------------
__global__ void proj_head(const float* __restrict__ h2,
                          const float* __restrict__ Wp, const float* __restrict__ bp,
                          const float* __restrict__ Wh1, const float* __restrict__ bh1,
                          const float* __restrict__ Wh2, const float* __restrict__ bh2,
                          float* __restrict__ emb, float* __restrict__ risk) {
    int node = (blockIdx.x * blockDim.x + threadIdx.x) >> 5;
    if (node >= NNODES) return;
    int lane = threadIdx.x & 31;
    int n = lane & 15, hf = lane >> 4;
    const float* hrow = h2 + (size_t)node * HC;
    float s = 0.f;
#pragma unroll 8
    for (int c = hf * 32; c < hf * 32 + 32; ++c) s += hrow[c] * Wp[c * OUTC + n];
    s += __shfl_xor(s, 16, 32);
    float embv = s + bp[n];
    if (lane < 16) emb[(size_t)node * OUTC + lane] = embv;
    // risk = relu(emb @ Wh1 + bh1) @ Wh2 + bh2
    int j = lane & 7;
    float t = bh1[j];
#pragma unroll
    for (int nn = 0; nn < 16; ++nn) t += __shfl(embv, nn, 32) * Wh1[nn * 8 + j];
    t = (t > 0.f) ? t : 0.f;
    float r = (lane < 8) ? t * Wh2[j] : 0.f;
#pragma unroll
    for (int o = 16; o; o >>= 1) r += __shfl_xor(r, o, 32);
    if (lane == 0) risk[node] = r + bh2[0];
}

// ======================================================================
extern "C" void kernel_launch(void* const* d_in, const int* in_sizes, int n_in,
                              void* d_out, int out_size, void* d_ws, size_t ws_size,
                              hipStream_t stream) {
    const float* x    = (const float*)d_in[0];
    const int*   ei   = (const int*)d_in[1];
    const float* Wl1  = (const float*)d_in[2];
    const float* bl1  = (const float*)d_in[3];
    const float* Wr1  = (const float*)d_in[4];
    const float* br1  = (const float*)d_in[5];
    const float* att1 = (const float*)d_in[6];
    const float* b1   = (const float*)d_in[7];
    const float* Wl2  = (const float*)d_in[8];
    const float* bl2  = (const float*)d_in[9];
    const float* Wr2  = (const float*)d_in[10];
    const float* br2  = (const float*)d_in[11];
    const float* att2 = (const float*)d_in[12];
    const float* b2   = (const float*)d_in[13];
    const float* bng  = (const float*)d_in[14];
    const float* bnb  = (const float*)d_in[15];
    const float* bnm  = (const float*)d_in[16];
    const float* bnv  = (const float*)d_in[17];
    const float* Wp   = (const float*)d_in[18];
    const float* bp   = (const float*)d_in[19];
    const float* Wh1  = (const float*)d_in[20];
    const float* bh1  = (const float*)d_in[21];
    const float* Wh2  = (const float*)d_in[22];
    const float* bh2  = (const float*)d_in[23];

    float* out_emb  = (float*)d_out;                         // [N,16]
    float* out_risk = (float*)d_out + (size_t)NNODES * OUTC; // [N,1]

    // ----- workspace carve-up (~72 MB) -----
    char* w = (char*)d_ws;
    _Float16* act16 = (_Float16*)w; w += (size_t)NNODES * 128 * sizeof(_Float16); // f16 acts
    float* xlxr  = (float*)w;   w += (size_t)NNODES * 128 * 4;   // [N][xl(64)|xr(64)]
    float* alpha = (float*)w;   w += (size_t)ETOT * 2 * 4;       // per (edge,head)
    unsigned* amax = (unsigned*)w; w += (size_t)NNODES * 2 * 4;  // keyed segment max
    float* denom = (float*)w;   w += (size_t)NNODES * 2 * 4;
    float* accum = (float*)w;   w += (size_t)NNODES * HC * 4;
    float* h2    = (float*)w;   w += (size_t)NNODES * HC * 4;
    _Float16* wpk1 = (_Float16*)w; w += 128 * 128 * sizeof(_Float16); // packed [Wl1|Wr1]
    _Float16* wpk2 = (_Float16*)w; w += 64 * 128 * sizeof(_Float16);  // packed [Wl2|Wr2]

    const int BLK = 256;
    const int edgeBlocks = (ETOT + 7) / 8;              // wave per edge (both heads)
    const int gemmBlocks = (NNODES + 127) / 128;
    const int nodeBlocks = (NNODES * HC + BLK - 1) / BLK;

    // weights -> WMMA fragment order
    pack_weights<<<(128 * 128 + BLK - 1) / BLK, BLK, 0, stream>>>(FIN, Wl1, Wr1, wpk1);
    pack_weights<<<(64 * 128 + BLK - 1) / BLK, BLK, 0, stream>>>(HC, Wl2, Wr2, wpk2);
    // x -> f16
    cvt_f16<<<(NNODES * FIN + BLK - 1) / BLK, BLK, 0, stream>>>(x, act16, NNODES * FIN);

    // ================= layer 1 =================
    init_layer<<<nodeBlocks, BLK, 0, stream>>>(accum, amax, denom);
    gemm_wmma<<<gemmBlocks, BLK, 0, stream>>>(act16, FIN, wpk1, bl1, br1, xlxr, NNODES);
    edge_alpha<<<edgeBlocks, BLK, 0, stream>>>(xlxr, ei, att1, alpha, amax);
    edge_accum<<<edgeBlocks, BLK, 0, stream>>>(xlxr, ei, alpha, amax, denom, accum);
    epilogue1<<<nodeBlocks, BLK, 0, stream>>>(accum, denom, b1, act16); // h1 (relu) -> f16

    // ================= layer 2 =================
    init_layer<<<nodeBlocks, BLK, 0, stream>>>(accum, amax, denom);
    gemm_wmma<<<gemmBlocks, BLK, 0, stream>>>(act16, HC, wpk2, bl2, br2, xlxr, NNODES);
    edge_alpha<<<edgeBlocks, BLK, 0, stream>>>(xlxr, ei, att2, alpha, amax);
    edge_accum<<<edgeBlocks, BLK, 0, stream>>>(xlxr, ei, alpha, amax, denom, accum);
    epilogue2<<<nodeBlocks, BLK, 0, stream>>>(accum, denom, b2, bng, bnb, bnm, bnv, h2);

    // ================= heads =================
    proj_head<<<(NNODES + 7) / 8, BLK, 0, stream>>>(h2, Wp, bp, Wh1, bh1, Wh2, bh2,
                                                    out_emb, out_risk);
}
